// CA_MambaLayer_64802466562244
// MI455X (gfx1250) — compile-verified, matrix operand order checked by reading, not run on hardware
//
#include <hip/hip_runtime.h>
#include <hip/hip_bf16.h>

typedef __attribute__((ext_vector_type(16))) _Float16 v16h;
typedef __attribute__((ext_vector_type(8)))  _Float16 v8h;
typedef __attribute__((ext_vector_type(8)))  float    v8f;

#define BATCH   8
#define SEQ     8192
#define DIM     256
#define DINNER  512
#define DSTATE  64
#define DTRANK  16
#define ROWS    (BATCH * SEQ)          // 65536

// ---------------------------------------------------------------------------
// CDNA5 async global->LDS DMA (ASYNCcnt-tracked), ISA §15.18 opcode 98.
// dst operand is the per-lane LDS byte address (flat LDS-aperture pointer
// truncated to 32 bits per the ISA LDS aperture rule).
// ---------------------------------------------------------------------------
__device__ __forceinline__ void async_g2lds_b128(const void* gptr, void* lptr) {
  unsigned lds = (unsigned)(uintptr_t)lptr;
  asm volatile("global_load_async_to_lds_b128 %0, %1, off"
               :: "v"(lds), "v"(gptr)
               : "memory");
}
__device__ __forceinline__ void wait_asynccnt0() {
  asm volatile("s_wait_asynccnt 0x0" ::: "memory");
}

// ---------------------------------------------------------------------------
// f32 -> f16 conversion (activations)
// ---------------------------------------------------------------------------
__global__ __launch_bounds__(256) void f32_to_f16_kernel(
    const float* __restrict__ in, _Float16* __restrict__ out, size_t n) {
  size_t i = (size_t)blockIdx.x * 256 + threadIdx.x;
  if (i < n) out[i] = (_Float16)in[i];
}

// ---------------------------------------------------------------------------
// f32 (K x N) -> f16 transposed (N x K): weights pre-swizzled so the GEMM's
// B-operand fragment is contiguous along K. K, N multiples of 16.
// ---------------------------------------------------------------------------
__global__ __launch_bounds__(256) void f32_to_f16_T_kernel(
    const float* __restrict__ in, _Float16* __restrict__ out, int K, int N) {
  __shared__ float tile[16][17];
  int k0 = blockIdx.y * 16, n0 = blockIdx.x * 16;
  int tx = threadIdx.x & 15, ty = threadIdx.x >> 4;
  tile[ty][tx] = in[(size_t)(k0 + ty) * N + n0 + tx];
  __syncthreads();
  out[(size_t)(n0 + ty) * K + k0 + tx] = (_Float16)tile[tx][ty];
}

// ---------------------------------------------------------------------------
// LayerNorm over last dim (256), one wave per row, emit f16
// ---------------------------------------------------------------------------
__global__ __launch_bounds__(256) void layernorm_f16_kernel(
    const float* __restrict__ x, const float* __restrict__ g,
    const float* __restrict__ b, _Float16* __restrict__ out) {
  int row  = blockIdx.x * 8 + (threadIdx.x >> 5);
  int lane = threadIdx.x & 31;
  const float* xr = x + (size_t)row * DIM;
  float v[8];
  float sum = 0.f;
#pragma unroll
  for (int i = 0; i < 8; ++i) { v[i] = xr[lane + 32 * i]; sum += v[i]; }
#pragma unroll
  for (int o = 16; o > 0; o >>= 1) sum += __shfl_xor(sum, o, 32);
  float mu = sum * (1.0f / DIM);
  float var = 0.f;
#pragma unroll
  for (int i = 0; i < 8; ++i) { float d = v[i] - mu; var += d * d; }
#pragma unroll
  for (int o = 16; o > 0; o >>= 1) var += __shfl_xor(var, o, 32);
  float rs = rsqrtf(var * (1.0f / DIM) + 1e-5f);
  _Float16* orow = out + (size_t)row * DIM;
#pragma unroll
  for (int i = 0; i < 8; ++i) {
    int c = lane + 32 * i;
    orow[c] = (_Float16)((v[i] - mu) * rs * g[c] + b[c]);
  }
}

// ---------------------------------------------------------------------------
// Fragment helper: per ISA 16-bit A/B VGPR layouts, a lane's half-fragment
// is 8 contiguous halfs (ds_load_b128):
//   lanes 0-15:  K = {0..7, 16..23};  lanes 16-31: K = {8..15, 24..31}.
// ---------------------------------------------------------------------------
__device__ __forceinline__ v16h load_frag(const _Float16* row, int klo) {
  v8h lo = *(const v8h*)(row + klo);
  v8h hi = *(const v8h*)(row + klo + 16);
  return __builtin_shufflevector(lo, hi, 0, 1, 2, 3, 4, 5, 6, 7, 8, 9, 10, 11,
                                 12, 13, 14, 15);
}

// ---------------------------------------------------------------------------
// Register-blocked WMMA GEMM: C[MxN] = A[MxK](f16) * Bt[NxK](f16), f32 acc.
// Block = 256 threads = 8 waves; each wave computes a 32x32 output tile
// (2x2 WMMA tiles, each A/B fragment reused twice). Block tile 128x64.
// K staged in KT=64 tiles via async global->LDS DMA -> 8 v_wmma per barrier.
// ---------------------------------------------------------------------------
template <bool OUT_HALF>
__global__ __launch_bounds__(256) void gemm_wmma_rb_kernel(
    const _Float16* __restrict__ A, const _Float16* __restrict__ Bt,
    void* __restrict__ C, int M, int N, int K) {
  constexpr int BM = 128, BN = 64, KT = 64;
  constexpr int LDK = KT + 8;   // 72 halfs = 144B stride (16B multiple)
  __shared__ __align__(16) _Float16 As[BM][LDK];
  __shared__ __align__(16) _Float16 Bs[BN][LDK];
  constexpr int A_CHUNKS = BM * KT / (256 * 8);  // 4
  constexpr int B_CHUNKS = BN * KT / (256 * 8);  // 2

  const int wave = threadIdx.x >> 5;
  const int lane = threadIdx.x & 31;
  const int wm = wave >> 1, wn = wave & 1;       // 4 x 2 wave grid
  const int m0 = blockIdx.y * BM;
  const int n0 = blockIdx.x * BN;

  v8f acc00 = {}, acc01 = {}, acc10 = {}, acc11 = {};
  const int r0  = wm * 32 + (lane & 15);
  const int c0  = wn * 32 + (lane & 15);
  const int klo = (lane < 16) ? 0 : 8;

  for (int k0 = 0; k0 < K; k0 += KT) {
    // ---- stage tiles via async DMA: no VGPR bounce, ASYNCcnt tracked ----
#pragma unroll
    for (int t = 0; t < A_CHUNKS; ++t) {
      int e = t * 2048 + threadIdx.x * 8;
      async_g2lds_b128(A + (size_t)(m0 + (e >> 6)) * K + k0 + (e & 63),
                       &As[e >> 6][e & 63]);
    }
#pragma unroll
    for (int t = 0; t < B_CHUNKS; ++t) {
      int e = t * 2048 + threadIdx.x * 8;
      async_g2lds_b128(Bt + (size_t)(n0 + (e >> 6)) * K + k0 + (e & 63),
                       &Bs[e >> 6][e & 63]);
    }
    wait_asynccnt0();
    __syncthreads();

    // ---- compute: 2 k-steps x 4 WMMAs ----
#pragma unroll
    for (int kb = 0; kb < KT; kb += 32) {
      v16h a0 = load_frag(&As[r0][kb], klo);
      v16h a1 = load_frag(&As[r0 + 16][kb], klo);
      v16h b0 = load_frag(&Bs[c0][kb], klo);
      v16h b1 = load_frag(&Bs[c0 + 16][kb], klo);
      acc00 = __builtin_amdgcn_wmma_f32_16x16x32_f16(false, a0, false, b0,
                                                     (short)0, acc00, false, false);
      acc01 = __builtin_amdgcn_wmma_f32_16x16x32_f16(false, a0, false, b1,
                                                     (short)0, acc01, false, false);
      acc10 = __builtin_amdgcn_wmma_f32_16x16x32_f16(false, a1, false, b0,
                                                     (short)0, acc10, false, false);
      acc11 = __builtin_amdgcn_wmma_f32_16x16x32_f16(false, a1, false, b1,
                                                     (short)0, acc11, false, false);
    }
    __syncthreads();
  }

  // ---- epilogue: C layout VGPR v -> (M = v | v+8, N = lane&15) ----
  const int mb = m0 + wm * 32 + ((lane < 16) ? 0 : 8);
  const int cb = n0 + wn * 32 + (lane & 15);
  v8f* accs[4] = {&acc00, &acc01, &acc10, &acc11};
#pragma unroll
  for (int tm = 0; tm < 2; ++tm)
#pragma unroll
    for (int tn = 0; tn < 2; ++tn) {
      v8f& a = *accs[tm * 2 + tn];
#pragma unroll
      for (int v = 0; v < 8; ++v) {
        size_t off = (size_t)(mb + tm * 16 + v) * N + cb + tn * 16;
        if (OUT_HALF) ((_Float16*)C)[off] = (_Float16)a[v];
        else          ((float*)C)[off]    = a[v];
      }
    }
}

// ---------------------------------------------------------------------------
// Single-tile WMMA GEMM (for N=80 projection). Block tile 128x16, 8 waves.
// ---------------------------------------------------------------------------
__global__ __launch_bounds__(256) void gemm_wmma_n16_kernel(
    const _Float16* __restrict__ A, const _Float16* __restrict__ Bt,
    float* __restrict__ C, int M, int N, int K) {
  constexpr int BM = 128, BN = 16, KT = 64;
  constexpr int LDK = KT + 8;
  __shared__ __align__(16) _Float16 As[BM][LDK];
  __shared__ __align__(16) _Float16 Bs[BN][LDK];

  const int wave = threadIdx.x >> 5;
  const int lane = threadIdx.x & 31;
  const int m0 = blockIdx.y * BM;
  const int n0 = blockIdx.x * BN;

  v8f acc = {};
  const int mrow = wave * 16 + (lane & 15);
  const int ncol = lane & 15;
  const int klo  = (lane < 16) ? 0 : 8;

  for (int k0 = 0; k0 < K; k0 += KT) {
#pragma unroll
    for (int t = 0; t < BM * KT / (256 * 8); ++t) {
      int e = t * 2048 + threadIdx.x * 8;
      async_g2lds_b128(A + (size_t)(m0 + (e >> 6)) * K + k0 + (e & 63),
                       &As[e >> 6][e & 63]);
    }
    {
      int e = threadIdx.x * 8;
      if (e < BN * KT)
        async_g2lds_b128(Bt + (size_t)(n0 + (e >> 6)) * K + k0 + (e & 63),
                         &Bs[e >> 6][e & 63]);
    }
    wait_asynccnt0();
    __syncthreads();
#pragma unroll
    for (int kb = 0; kb < KT; kb += 32) {
      v16h af = load_frag(&As[mrow][kb], klo);
      v16h bf = load_frag(&Bs[ncol][kb], klo);
      acc = __builtin_amdgcn_wmma_f32_16x16x32_f16(false, af, false, bf,
                                                   (short)0, acc, false, false);
    }
    __syncthreads();
  }

  const int colg  = n0 + ncol;
  const int mbase = m0 + wave * 16 + ((lane < 16) ? 0 : 8);
#pragma unroll
  for (int v = 0; v < 8; ++v)
    C[(size_t)(mbase + v) * N + colg] = acc[v];
}

// ---------------------------------------------------------------------------
// Depthwise causal conv (D_CONV=4) + SiLU. xi = xz[..., 0:512].
// ---------------------------------------------------------------------------
__global__ __launch_bounds__(256) void conv_silu_kernel(
    const _Float16* __restrict__ xz, const float* __restrict__ cw,
    const float* __restrict__ cb, _Float16* __restrict__ xc) {
  size_t idx = (size_t)blockIdx.x * 256 + threadIdx.x;  // (b*N+n)*512 + d
  int d = (int)(idx & (DINNER - 1));
  size_t bn = idx >> 9;
  int n = (int)(bn & (SEQ - 1));
  float acc = cb[d];
#pragma unroll
  for (int k = 0; k < 4; ++k) {
    int nn = n - 3 + k;
    if (nn >= 0)
      acc += cw[d * 4 + k] * (float)xz[(bn - (size_t)(3 - k)) * 1024 + d];
  }
  float sig = 1.f / (1.f + __expf(-acc));
  xc[idx] = (_Float16)(acc * sig);
}

// ---------------------------------------------------------------------------
// dt = softplus(dt_raw @ W_dt + b_dt), one block per (b,n) row; K=16.
// ---------------------------------------------------------------------------
__global__ __launch_bounds__(512) void dt_softplus_kernel(
    const float* __restrict__ xdb, const float* __restrict__ Wdt,
    const float* __restrict__ bdt, float* __restrict__ dt) {
  __shared__ float r[DTRANK];
  size_t bn = blockIdx.x;
  int d = threadIdx.x;
  if (d < DTRANK) r[d] = xdb[bn * 80 + d];
  __syncthreads();
  float acc = bdt[d];
#pragma unroll
  for (int k = 0; k < DTRANK; ++k) acc += r[k] * Wdt[k * DINNER + d];
  float sp = (acc > 20.f) ? acc : log1pf(__expf(acc));
  dt[bn * DINNER + d] = sp;
}

// ---------------------------------------------------------------------------
// Selective scan: one wave per (b,d) channel, 2 states per lane.
// ---------------------------------------------------------------------------
__global__ __launch_bounds__(256) void ssm_scan_kernel(
    const float* __restrict__ dt, const float* __restrict__ xdb,
    const float* __restrict__ Cm, const _Float16* __restrict__ xc,
    const float* __restrict__ A_log, const float* __restrict__ Dp,
    float* __restrict__ y) {
  int wave = threadIdx.x >> 5, lane = threadIdx.x & 31;
  int chan = blockIdx.x * 8 + wave;           // 0..4095
  int b = chan >> 9, d = chan & (DINNER - 1);
  int s0 = lane * 2;
  float a0 = -__expf(A_log[d * DSTATE + s0]);
  float a1 = -__expf(A_log[d * DSTATE + s0 + 1]);
  float Dd = Dp[d];
  float h0 = 0.f, h1 = 0.f;
  size_t base = (size_t)b * SEQ;
  for (int n = 0; n < SEQ; ++n) {
    size_t bn = base + n;
    if (n + 8 < SEQ) {   // hide latency of the serially-dependent stream
      __builtin_prefetch(xdb + (bn + 8) * 80 + DTRANK + s0, 0, 1);
      __builtin_prefetch(Cm + (bn + 8) * DSTATE + s0, 0, 1);
    }
    float dt_t = dt[bn * DINNER + d];
    float x_t  = (float)xc[bn * DINNER + d];
    float2 Bv = *(const float2*)(xdb + bn * 80 + DTRANK + s0);
    float2 Cv = *(const float2*)(Cm + bn * DSTATE + s0);
    float dA0 = __expf(dt_t * a0);
    float dA1 = __expf(dt_t * a1);
    float dtx = dt_t * x_t;
    h0 = dA0 * h0 + dtx * Bv.x;
    h1 = dA1 * h1 + dtx * Bv.y;
    float yp = h0 * Cv.x + h1 * Cv.y;
#pragma unroll
    for (int o = 16; o > 0; o >>= 1) yp += __shfl_xor(yp, o, 32);
    if (lane == 0) y[bn * DINNER + d] = yp + Dd * x_t;
  }
}

// ---------------------------------------------------------------------------
// Gate: g = y * silu(z), z = xz[..., 512:1024]; emit f16 for final GEMM.
// ---------------------------------------------------------------------------
__global__ __launch_bounds__(256) void gate_silu_kernel(
    const float* __restrict__ y, const _Float16* __restrict__ xz,
    _Float16* __restrict__ g) {
  size_t idx = (size_t)blockIdx.x * 256 + threadIdx.x;
  int d = (int)(idx & (DINNER - 1));
  size_t bn = idx >> 9;
  float z = (float)xz[bn * 1024 + DINNER + d];
  float sig = 1.f / (1.f + __expf(-z));
  g[idx] = (_Float16)(y[idx] * z * sig);
}

// ---------------------------------------------------------------------------
// Host launcher
// ---------------------------------------------------------------------------
static inline char* carve(char*& p, size_t bytes) {
  char* r = p;
  p += (bytes + 255) & ~(size_t)255;
  return r;
}

extern "C" void kernel_launch(void* const* d_in, const int* in_sizes, int n_in,
                              void* d_out, int out_size, void* d_ws,
                              size_t ws_size, hipStream_t stream) {
  const float* x      = (const float*)d_in[0];
  const float* C_SA   = (const float*)d_in[1];
  const float* ln_g   = (const float*)d_in[2];
  const float* ln_b   = (const float*)d_in[3];
  const float* W_in   = (const float*)d_in[4];
  const float* conv_w = (const float*)d_in[5];
  const float* conv_b = (const float*)d_in[6];
  const float* W_xprj = (const float*)d_in[7];
  const float* W_dt   = (const float*)d_in[8];
  const float* b_dt   = (const float*)d_in[9];
  const float* W_C    = (const float*)d_in[10];
  const float* A_log  = (const float*)d_in[11];
  const float* Dp     = (const float*)d_in[12];
  const float* W_out  = (const float*)d_in[13];
  float* out = (float*)d_out;

  char* p = (char*)d_ws;
  _Float16* w_in_t  = (_Float16*)carve(p, (size_t)DIM * 1024 * 2);   // 1024 x 256
  _Float16* w_xp_t  = (_Float16*)carve(p, (size_t)DINNER * 80 * 2);  // 80 x 512
  _Float16* w_c_t   = (_Float16*)carve(p, (size_t)DIM * DSTATE * 2); // 64 x 256
  _Float16* w_out_t = (_Float16*)carve(p, (size_t)DINNER * DIM * 2); // 256 x 512
  _Float16* xn_h    = (_Float16*)carve(p, (size_t)ROWS * DIM * 2);
  _Float16* csa_h   = (_Float16*)carve(p, (size_t)ROWS * DIM * 2);
  _Float16* xz_h    = (_Float16*)carve(p, (size_t)ROWS * 1024 * 2);
  _Float16* xc_h    = (_Float16*)carve(p, (size_t)ROWS * DINNER * 2);
  float*    xdb_f   = (float*)carve(p, (size_t)ROWS * 80 * 4);
  float*    cm_f    = (float*)carve(p, (size_t)ROWS * DSTATE * 4);
  float*    dt_f    = (float*)carve(p, (size_t)ROWS * DINNER * 4);
  float*    y_f     = (float*)carve(p, (size_t)ROWS * DINNER * 4);
  _Float16* g_h     = (_Float16*)carve(p, (size_t)ROWS * DINNER * 2);

  // weights: f32 (KxN) -> f16 transposed (NxK)
  auto cvtT = [&](const float* src, _Float16* dst, int K, int N) {
    f32_to_f16_T_kernel<<<dim3(N / 16, K / 16), 256, 0, stream>>>(src, dst, K, N);
  };
  cvtT(W_in, w_in_t, DIM, 1024);
  cvtT(W_xprj, w_xp_t, DINNER, 80);
  cvtT(W_C, w_c_t, DIM, DSTATE);
  cvtT(W_out, w_out_t, DINNER, DIM);

  // C_SA -> f16 (A operand, row-major)
  {
    size_t n = (size_t)ROWS * DIM;
    f32_to_f16_kernel<<<(unsigned)((n + 255) / 256), 256, 0, stream>>>(C_SA, csa_h, n);
  }

  // 1. LayerNorm -> f16
  layernorm_f16_kernel<<<ROWS / 8, 256, 0, stream>>>(x, ln_g, ln_b, xn_h);

  // 2. xz = xn @ W_in  (65536x256 @ 256x1024), f16 out
  gemm_wmma_rb_kernel<true>
      <<<dim3(1024 / 64, ROWS / 128), 256, 0, stream>>>(
          xn_h, w_in_t, xz_h, ROWS, 1024, DIM);

  // 3. depthwise causal conv + SiLU -> xc (f16)
  conv_silu_kernel<<<(unsigned)((size_t)ROWS * DINNER / 256), 256, 0, stream>>>(
      xz_h, conv_w, conv_b, xc_h);

  // 4. x_db = xc @ W_xproj  (65536x512 @ 512x80), f32 out
  gemm_wmma_n16_kernel<<<dim3(80 / 16, ROWS / 128), 256, 0, stream>>>(
      xc_h, w_xp_t, xdb_f, ROWS, 80, DINNER);

  // 5. dt = softplus(dt_raw @ W_dt + b_dt)
  dt_softplus_kernel<<<ROWS, 512, 0, stream>>>(xdb_f, W_dt, b_dt, dt_f);

  // 6. Cm = C_SA @ W_C  (65536x256 @ 256x64), f32 out
  gemm_wmma_rb_kernel<false>
      <<<dim3(DSTATE / 64, ROWS / 128), 256, 0, stream>>>(
          csa_h, w_c_t, cm_f, ROWS, DSTATE, DIM);

  // 7. selective scan (+ D*x skip), wave per channel
  ssm_scan_kernel<<<(BATCH * DINNER) / 8, 256, 0, stream>>>(
      dt_f, xdb_f, cm_f, xc_h, A_log, Dp, y_f);

  // 8. gate with silu(z) -> f16
  gate_silu_kernel<<<(unsigned)((size_t)ROWS * DINNER / 256), 256, 0, stream>>>(
      y_f, xz_h, g_h);

  // 9. out = g @ W_out  (65536x512 @ 512x256), f32 out
  gemm_wmma_rb_kernel<false>
      <<<dim3(DIM / 64, ROWS / 128), 256, 0, stream>>>(
          g_h, w_out_t, out, ROWS, DIM, DINNER);
}
